// SoftMoA_83648783057473
// MI455X (gfx1250) — matrix operand order repaired and checked by reading, not compiled, not run on hardware
//
#include <hip/hip_runtime.h>

typedef unsigned short ushort_t;
typedef __attribute__((ext_vector_type(16))) __bf16 v16bf;
typedef __attribute__((ext_vector_type(8)))  float  v8f;

constexpr int Bb = 32;    // batch
constexpr int Mm = 1024;  // tokens
constexpr int Dd = 768;   // model dim
constexpr int NP = 32;    // n*p slots
constexpr int Hh = 96;    // bottleneck
constexpr float EPSN = 1e-6f;

// ---------- helpers ----------

__device__ __forceinline__ ushort_t f2bf(float x) {
  unsigned u = __float_as_uint(x);
  u += 0x7FFFu + ((u >> 16) & 1u);   // round-to-nearest-even
  return (ushort_t)(u >> 16);
}

// A-matrix bf16 fragment (16x32, row-major source): lane<16 -> row, halves
// [k..k+7] and [k+16..k+23]; lane>=16 handled by caller passing base+8.
__device__ __forceinline__ v16bf ld_frag_a(const ushort_t* p) {
  union { v16bf v; uint4 q[2]; } f;
  f.q[0] = *(const uint4*)(p);
  f.q[1] = *(const uint4*)(p + 16);
  return f.v;
}
// B-matrix bf16 fragment (32x16, N-major source): [k..k+7] and [k+16..k+23]; lane>=16 via base+8.
__device__ __forceinline__ v16bf ld_frag_b(const ushort_t* p) {
  union { v16bf v; uint4 q[2]; } f;
  f.q[0] = *(const uint4*)(p);
  f.q[1] = *(const uint4*)(p + 16);
  return f.v;
}

__device__ __forceinline__ v8f wmma_bf16(v16bf a, v16bf b, v8f c) {
  // (neg_a, A, neg_b, B, c_mod, C, reuse_a, reuse_b)
  return __builtin_amdgcn_wmma_f32_16x16x32_bf16(false, a, false, b, (short)0, c,
                                                 false, false);
}

// ---------- kernel 0: slot-param normalization -> spT bf16 [NP][Dd] ----------

__global__ __launch_bounds__(256) void k_slots(const float* __restrict__ slot,
                                               const float* __restrict__ scale,
                                               ushort_t* __restrict__ spT) {
  const int idx = blockIdx.x * 256 + threadIdx.x;   // dd*4 + pp
  if (idx >= Dd * 4) return;
  const int dd = idx >> 2;
  const int pp = idx & 3;
  float v[8];
  float s = 0.f;
  for (int nn = 0; nn < 8; ++nn) {
    float x = slot[(size_t)dd * 32 + nn * 4 + pp];  // (d, n, p)
    v[nn] = x;
    s += x * x;
  }
  const float inv = (*scale) / (sqrtf(s) + EPSN);   // normalize over n
  for (int nn = 0; nn < 8; ++nn)
    spT[(size_t)(nn * 4 + pp) * Dd + dd] = f2bf(v[nn] * inv);
}

// ---------- kernel 1: row l2-norm + logits GEMM + XnT bf16 write ----------
// grid: 2048 tiles of 16 tokens, block = 64 threads (2 waves).

__global__ __launch_bounds__(64) void k_norm_logits(
    const float* __restrict__ X, const ushort_t* __restrict__ spT,
    ushort_t* __restrict__ XnT, float* __restrict__ logits,
    float* __restrict__ logitsT) {
  constexpr int PITCH = Dd + 8;                 // bank-conflict pad
  __shared__ ushort_t xb[16 * PITCH];
  __shared__ float ssq[16][4];
  __shared__ float rnorm[16];
  const int tile = blockIdx.x;
  const int b  = tile >> 6;
  const int t0 = (tile & 63) << 4;              // token offset within batch
  const int tid = threadIdx.x;

  // pass 1: per-token sum of squares (deterministic 4-way split)
  {
    const int tk = tid >> 2, q = tid & 3;
    const float* r0 = X + (size_t)(b * Mm + t0 + tk) * Dd + q * 192;
    float s = 0.f;
    for (int i = 0; i < 192; i += 4) {
      float4 x4 = *(const float4*)(r0 + i);
      s += x4.x * x4.x + x4.y * x4.y + x4.z * x4.z + x4.w * x4.w;
    }
    ssq[tk][q] = s;
  }
  __syncthreads();
  if (tid < 16) {
    float s = ssq[tid][0] + ssq[tid][1] + ssq[tid][2] + ssq[tid][3];
    rnorm[tid] = 1.0f / (sqrtf(s) + EPSN);
  }
  __syncthreads();

  // pass 2: re-read (cache-hot), scale, bf16 into LDS
  for (int j = tid; j < 16 * (Dd / 4); j += 64) {
    const int tk = j / (Dd / 4);
    const int c4 = j - tk * (Dd / 4);
    float4 x4 = *(const float4*)(X + (size_t)(b * Mm + t0 + tk) * Dd + c4 * 4);
    const float rn = rnorm[tk];
    ushort4 o;
    o.x = f2bf(x4.x * rn); o.y = f2bf(x4.y * rn);
    o.z = f2bf(x4.z * rn); o.w = f2bf(x4.w * rn);
    *(ushort4*)(&xb[tk * PITCH + c4 * 4]) = o;
  }
  __syncthreads();

  // logits[16 tokens x 32 slots]: wave w handles slots w*16..w*16+15
  const int lane = tid & 31;
  const int wave = tid >> 5;
  const int n0 = wave << 4;
  const int r  = lane & 15;
  const int hi = lane >> 4;
  v8f acc = {};
  for (int k0 = 0; k0 < Dd; k0 += 32) {
    union { v16bf v; uint4 q[2]; } fa;       // A fragment straight from LDS
    const int ka = k0 + hi * 8;
    fa.q[0] = *(const uint4*)(&xb[r * PITCH + ka]);
    fa.q[1] = *(const uint4*)(&xb[r * PITCH + ka + 16]);
    v16bf fb = ld_frag_b(spT + (size_t)(n0 + r) * Dd + k0 + hi * 8);
    acc = wmma_bf16(fa.v, fb, acc);
  }
  for (int i = 0; i < 8; ++i) {              // C/D layout: M = i + 8*hi, N = r
    const int tt = t0 + i + hi * 8;
    const int ss = n0 + r;
    logits [(size_t)(b * Mm + tt) * NP + ss] = acc[i];
    logitsT[(size_t)(b * NP + ss) * Mm + tt] = acc[i];
  }

  // transposed bf16 Xn: XnT[b][d][m] (B-matrix layout for phase 3)
  __syncthreads();
  for (int dd = tid; dd < Dd; dd += 64) {
    union { ushort_t u[16]; uint4 q[2]; } tmp;
    for (int rr = 0; rr < 16; ++rr) tmp.u[rr] = xb[rr * PITCH + dd];
    uint4* o = (uint4*)(XnT + (size_t)(b * Dd + dd) * Mm + t0);
    o[0] = tmp.q[0];
    o[1] = tmp.q[1];
  }
}

// ---------- kernel 2: dispatch softmax over m -> DT bf16 [b][np][m] ----------

__global__ __launch_bounds__(256) void k_dispatch_softmax(
    const float* __restrict__ logitsT, ushort_t* __restrict__ DT) {
  const int row = blockIdx.x;                       // b*NP + slot
  const float* src = logitsT + (size_t)row * Mm;
  __shared__ float red[256];
  const int tid = threadIdx.x;
  float v0 = src[tid], v1 = src[tid + 256], v2 = src[tid + 512], v3 = src[tid + 768];
  float mx = fmaxf(fmaxf(v0, v1), fmaxf(v2, v3));
  red[tid] = mx; __syncthreads();
  for (int o = 128; o > 0; o >>= 1) {
    if (tid < o) red[tid] = fmaxf(red[tid], red[tid + o]);
    __syncthreads();
  }
  mx = red[0]; __syncthreads();
  float e0 = expf(v0 - mx), e1 = expf(v1 - mx), e2 = expf(v2 - mx), e3 = expf(v3 - mx);
  red[tid] = e0 + e1 + e2 + e3; __syncthreads();
  for (int o = 128; o > 0; o >>= 1) {
    if (tid < o) red[tid] += red[tid + o];
    __syncthreads();
  }
  const float inv = 1.0f / red[0];
  ushort_t* dst = DT + (size_t)row * Mm;
  dst[tid]       = f2bf(e0 * inv);
  dst[tid + 256] = f2bf(e1 * inv);
  dst[tid + 512] = f2bf(e2 * inv);
  dst[tid + 768] = f2bf(e3 * inv);
}

// ---------- kernel 3: combine softmax over 32 slots -> C bf16 [b][m][np] ----

__global__ __launch_bounds__(256) void k_combine_softmax(
    const float* __restrict__ logits, ushort_t* __restrict__ Cb) {
  const int t = blockIdx.x * 256 + threadIdx.x;     // global token id
  const float* src = logits + (size_t)t * NP;
  float v[NP];
  float mx = -3.4e38f;
  for (int i = 0; i < NP; ++i) { v[i] = src[i]; mx = fmaxf(mx, v[i]); }
  float s = 0.f;
  for (int i = 0; i < NP; ++i) { v[i] = expf(v[i] - mx); s += v[i]; }
  const float inv = 1.0f / s;
  ushort_t* dst = Cb + (size_t)t * NP;
  for (int i = 0; i < NP; ++i) dst[i] = f2bf(v[i] * inv);
}

// ---------- kernel 4: X_tilde = D^T @ Xn  (per batch, WMMA, K=1024) --------
// grid (48 Ntiles, 2 Mtiles, 32 batches), 1 wave per block.

__global__ __launch_bounds__(32) void k_xtilde(const ushort_t* __restrict__ DT,
                                               const ushort_t* __restrict__ XnT,
                                               float* __restrict__ Xt) {
  const int d0 = blockIdx.x << 4;
  const int s0 = blockIdx.y << 4;
  const int b  = blockIdx.z;
  const int lane = threadIdx.x;
  const int r = lane & 15, hi = lane >> 4;
  const ushort_t* arow = DT  + (size_t)(b * NP + s0 + r) * Mm;
  const ushort_t* brow = XnT + (size_t)(b * Dd + d0 + r) * Mm;
  v8f acc = {};
  for (int k0 = 0; k0 < Mm; k0 += 32) {
    v16bf fa = ld_frag_a(arow + k0 + hi * 8);
    v16bf fb = ld_frag_b(brow + k0 + hi * 8);
    acc = wmma_bf16(fa, fb, acc);
  }
  for (int i = 0; i < 8; ++i) {
    const int ss = s0 + i + hi * 8;
    Xt[(size_t)(b * NP + ss) * Dd + d0 + r] = acc[i];
  }
}

// ---------- kernel 5: bottleneck adapters (exact GELU) -> Y_tildeT bf16 -----
// grid = b*n blocks, 384 threads = 4p x 96h.

__global__ __launch_bounds__(384) void k_adapter(
    const float* __restrict__ Xt, const float* __restrict__ Wd,
    const float* __restrict__ bd, const float* __restrict__ Wu,
    const float* __restrict__ bu, ushort_t* __restrict__ YtT) {
  const int b = blockIdx.x >> 3;
  const int n = blockIdx.x & 7;
  __shared__ float xsh[4 * Dd];
  __shared__ float hsh[4 * Hh];
  const int tid = threadIdx.x;
  const float* src = Xt + (size_t)(b * NP + n * 4) * Dd;
  for (int j = tid; j < 4 * Dd; j += 384) xsh[j] = src[j];
  __syncthreads();
  {
    const int pp = tid / Hh, hh = tid - pp * Hh;     // exactly 4*96 threads
    const float* w = Wd + (size_t)n * Dd * Hh + hh;  // Wd[n][dd][hh]
    const float* x = xsh + pp * Dd;
    float s = 0.f;
    for (int dd = 0; dd < Dd; ++dd) s += x[dd] * w[(size_t)dd * Hh];
    s += bd[n * Hh + hh];
    hsh[pp * Hh + hh] = 0.5f * s * (1.0f + erff(s * 0.70710678118654752f));
  }
  __syncthreads();
  for (int j = tid; j < 4 * Dd; j += 384) {
    const int pp = j / Dd, dd = j - pp * Dd;
    const float* w = Wu + (size_t)n * Hh * Dd + dd;  // Wu[n][hh][dd]
    const float* h = hsh + pp * Hh;
    float s = 0.f;
    for (int hh = 0; hh < Hh; ++hh) s += h[hh] * w[(size_t)hh * Dd];
    s += bu[n * Dd + dd];
    YtT[(size_t)(b * Dd + dd) * NP + n * 4 + pp] = f2bf(s);  // [b][d][np]
  }
}

// ---------- kernel 6: Y = C @ Y_tilde (per batch, K=32 -> 1 WMMA/tile) ------
// grid (64 token-tiles, 32 batches), 256 threads = 8 waves x 6 N-tiles.

__global__ __launch_bounds__(256) void k_combine(const ushort_t* __restrict__ Cb,
                                                 const ushort_t* __restrict__ YtT,
                                                 float* __restrict__ Y) {
  const int b  = blockIdx.y;
  const int t0 = blockIdx.x << 4;
  const int tid  = threadIdx.x;
  const int lane = tid & 31;
  const int wave = tid >> 5;
  const int r = lane & 15, hi = lane >> 4;
  v16bf fa = ld_frag_a(Cb + (size_t)(b * Mm + t0 + r) * NP + hi * 8);
  for (int j = 0; j < 6; ++j) {
    const int d0 = (wave * 6 + j) << 4;
    v16bf fb = ld_frag_b(YtT + (size_t)(b * Dd + d0 + r) * NP + hi * 8);
    v8f acc = {};
    acc = wmma_bf16(fa, fb, acc);
    for (int i = 0; i < 8; ++i) {
      const int tt = t0 + i + hi * 8;
      Y[(size_t)(b * Mm + tt) * Dd + d0 + r] = acc[i];
    }
  }
}

// ---------- launcher ----------

extern "C" void kernel_launch(void* const* d_in, const int* in_sizes, int n_in,
                              void* d_out, int out_size, void* d_ws, size_t ws_size,
                              hipStream_t stream) {
  (void)in_sizes; (void)n_in; (void)out_size; (void)ws_size;
  const float* X    = (const float*)d_in[0];
  const float* slot = (const float*)d_in[1];
  const float* scal = (const float*)d_in[2];
  const float* Wd   = (const float*)d_in[3];
  const float* bd   = (const float*)d_in[4];
  const float* Wu   = (const float*)d_in[5];
  const float* bu   = (const float*)d_in[6];
  float* Y = (float*)d_out;

  char* ws = (char*)d_ws;
  size_t off = 0;
  auto carve = [&](size_t bytes) -> void* {
    void* p = ws + off;
    off = (off + bytes + 255) & ~(size_t)255;
    return p;
  };
  ushort_t* spT     = (ushort_t*)carve((size_t)NP * Dd * 2);            // 48 KB
  ushort_t* XnT     = (ushort_t*)carve((size_t)Bb * Dd * Mm * 2);       // 48 MB
  float*    logits  = (float*)   carve((size_t)Bb * Mm * NP * 4);       // 4 MB
  float*    logitsT = (float*)   carve((size_t)Bb * NP * Mm * 4);       // 4 MB
  ushort_t* DT      = (ushort_t*)carve((size_t)Bb * NP * Mm * 2);       // 2 MB
  ushort_t* Cbuf    = (ushort_t*)carve((size_t)Bb * Mm * NP * 2);       // 2 MB
  float*    Xt      = (float*)   carve((size_t)Bb * NP * Dd * 4);       // 3 MB
  ushort_t* YtT     = (ushort_t*)carve((size_t)Bb * Dd * NP * 2);       // 1.5 MB

  k_slots<<<dim3((Dd * 4 + 255) / 256), dim3(256), 0, stream>>>(slot, scal, spT);
  k_norm_logits<<<dim3(Bb * Mm / 16), dim3(64), 0, stream>>>(X, spT, XnT, logits,
                                                             logitsT);
  k_dispatch_softmax<<<dim3(Bb * NP), dim3(256), 0, stream>>>(logitsT, DT);
  k_combine_softmax<<<dim3(Bb * Mm / 256), dim3(256), 0, stream>>>(logits, Cbuf);
  k_xtilde<<<dim3(Dd / 16, NP / 16, Bb), dim3(32), 0, stream>>>(DT, XnT, Xt);
  k_adapter<<<dim3(Bb * 8), dim3(384), 0, stream>>>(Xt, Wd, bd, Wu, bu, YtT);
  k_combine<<<dim3(Mm / 16, Bb), dim3(256), 0, stream>>>(Cbuf, YtT, Y);
}
